// GraphTokenEncoder_30562987278951
// MI455X (gfx1250) — compile-verified, matrix operand order checked by reading, not run on hardware
//
#include <hip/hip_runtime.h>
#include <hip/hip_bf16.h>
#include <math.h>

// ---------------------------------------------------------------------------
// GINE GNN for MI455X (gfx1250, wave32, WMMA + Tensor Data Mover).
//  - GEMMs: f16 A/B, f32 accumulate via v_wmma_f32_16x16x32_f16.
//  - Weight panel (128 cols x 512 K, f16, padded) loaded ONCE per block into
//    the 320KB WGP LDS with one tensor_load_to_lds; A tiles double-buffer
//    through TDM overlapped with WMMA compute (TENSORcnt).
//  - Edge scatter uses native global_atomic_add_f32 (unsafeAtomicAdd).
//  - h (node features, fp32) lives in d_out across layers.
// ---------------------------------------------------------------------------

typedef __attribute__((ext_vector_type(16))) _Float16 v16h;
typedef __attribute__((ext_vector_type(8)))  _Float16 v8h;
typedef __attribute__((ext_vector_type(8)))  float    v8f;
typedef __attribute__((ext_vector_type(4)))  unsigned int v4u;
typedef __attribute__((ext_vector_type(8)))  int      v8i;
typedef __attribute__((ext_vector_type(4)))  int      v4i;

union Frag16 { v16h v; v8h h[2]; };

static __device__ __forceinline__ v8f zero_v8f() {
    v8f a;
#pragma unroll
    for (int i = 0; i < 8; ++i) a[i] = 0.0f;
    return a;
}
static __device__ __forceinline__ float gelu_exact(float x) {
    return 0.5f * x * (1.0f + erff(x * 0.70710678118654752f));
}

#define DHID 512
#define BM 128
#define BN 128
#define BK 32
#define APAD 40    // A-tile LDS row stride in f16 (32 data + 8 pad, 80B -> 16B aligned)
#define BPAD 520   // B-panel LDS row stride in f16 (512 data + 8 pad, 1040B -> banks rotate)

// ---------------------------------------------------------------------------
// TDM 2D tile load: global (row-major, row_stride elems of f16) -> LDS with
// per-row padding.  D# packing per CDNA5 ISA.
// pad_interval_code: 0=2,1=4,2=8,3=16,...,7=256 DWORDs between pads
// pad_amount_code:   n -> (n+1) DWORDs of pad
// NOTE: this toolchain's builtin is the 6-arg form:
//       (v4u g0, v8i g1, v4i g2, v4i g3, v8i g4, i32 cpol)
// ---------------------------------------------------------------------------
static __device__ __forceinline__ void tdm_load_2d(unsigned lds_addr, const void* gaddr,
                                                   unsigned tile_k, unsigned tile_rows,
                                                   unsigned tensor_k, unsigned tensor_rows,
                                                   unsigned row_stride,
                                                   unsigned pad_interval_code,
                                                   unsigned pad_amount_code) {
    unsigned long long ga = (unsigned long long)gaddr;
    v4u g0;
    g0[0] = 1u;                                              // count=1, no gather
    g0[1] = lds_addr;                                        // lds_addr (bytes)
    g0[2] = (unsigned)(ga & 0xffffffffu);                    // global_addr[31:0]
    g0[3] = (unsigned)((ga >> 32) & 0x01ffffffu) | 0x80000000u;  // global_addr[56:32] | type=2
    v8i g1;
    g1[0] = (int)(0x00010000u                                // data_size=1 (2 bytes)
                  | (1u << 20)                               // pad_enable
                  | (pad_interval_code << 22)
                  | (pad_amount_code << 25));
    g1[1] = (int)(tensor_k << 16);                           // tensor_dim0[15:0]
    g1[2] = (int)((tensor_k >> 16) | (tensor_rows << 16));   // tensor_dim0[31:16] | tensor_dim1[15:0]
    g1[3] = (int)((tensor_rows >> 16) | (tile_k << 16));     // tensor_dim1[31:16] | tile_dim0
    g1[4] = (int)(tile_rows & 0xffffu);                      // tile_dim1 (tile_dim2=0)
    g1[5] = (int)row_stride;                                 // tensor_dim0_stride[31:0]
    g1[6] = 0;                                               // stride[47:32] | dim1_stride[15:0]
    g1[7] = 0;
    v4i gz;
    gz[0] = 0; gz[1] = 0; gz[2] = 0; gz[3] = 0;              // groups 2/3: zero (2D tile)
    v8i gz8;
#pragma unroll
    for (int i = 0; i < 8; ++i) gz8[i] = 0;
    __builtin_amdgcn_tensor_load_to_lds(g0, g1, gz, gz, gz8, 0);
}

// ---------------------------------------------------------------------------
// Weight transpose + f16 convert: W[k][n] (fp32, 512x512) -> Wt[n][k] (f16)
// ---------------------------------------------------------------------------
__global__ __launch_bounds__(256) void transpose_w_f16(const float* __restrict__ W,
                                                       _Float16* __restrict__ Wt) {
    int i = blockIdx.x * 256 + threadIdx.x;
    int k = i >> 9;
    int n = i & 511;
    Wt[(size_t)n * DHID + k] = (_Float16)W[(size_t)k * DHID + n];
}

// ---------------------------------------------------------------------------
// Embedding sum + LayerNorm -> f16 rows. One block (256 thr) per row.
// ---------------------------------------------------------------------------
__global__ __launch_bounds__(256) void embed_ln_f16(const int* __restrict__ idx,
                                                    const float* __restrict__ emb,
                                                    const float* __restrict__ g,
                                                    const float* __restrict__ b,
                                                    _Float16* __restrict__ out,
                                                    int F, int V) {
    const int row = blockIdx.x;
    const int t = threadIdx.x;
    float x0 = 0.0f, x1 = 0.0f;
    for (int f = 0; f < F; ++f) {
        int e = idx[row * F + f];
        const float* p = emb + ((size_t)f * V + e) * DHID;
        x0 += p[t];
        x1 += p[t + 256];
    }
    __shared__ float red[256];
    red[t] = x0 + x1;
    __syncthreads();
    for (int o = 128; o > 0; o >>= 1) {
        if (t < o) red[t] += red[t + o];
        __syncthreads();
    }
    const float mean = red[0] * (1.0f / 512.0f);
    __syncthreads();
    const float d0 = x0 - mean, d1 = x1 - mean;
    red[t] = d0 * d0 + d1 * d1;
    __syncthreads();
    for (int o = 128; o > 0; o >>= 1) {
        if (t < o) red[t] += red[t + o];
        __syncthreads();
    }
    const float rstd = rsqrtf(red[0] * (1.0f / 512.0f) + 1e-5f);
    size_t base = (size_t)row * DHID;
    out[base + t]       = (_Float16)(d0 * rstd * g[t] + b[t]);
    out[base + t + 256] = (_Float16)(d1 * rstd * g[t + 256] + b[t + 256]);
}

// ---------------------------------------------------------------------------
// WMMA GEMM: Y[M,512] = act(X[M,512] @ W[512,512] + bias)
//   Dynamic LDS: B panel [128][520] f16 (whole K, loaded once by TDM) +
//                double-buffered A tiles [2][128][40] f16 (TDM per K-step).
//   8 waves as 2(M)x4(N); wave owns 64x32 = 8 accumulators.
//   ACT: 0=none, 1=relu, 2=gelu.  OUT_F16: output dtype.
// ---------------------------------------------------------------------------
template <int ACT, bool OUT_F16>
__global__ __launch_bounds__(256) void gemm_wmma(const _Float16* __restrict__ X,
                                                 const _Float16* __restrict__ Wt,
                                                 const float* __restrict__ bias,
                                                 void* __restrict__ Yout, int M) {
    extern __shared__ __align__(16) _Float16 lds[];
    _Float16* Bp = lds;                       // [BN][BPAD]
    _Float16* At = lds + (size_t)BN * BPAD;   // [2][BM][APAD]

    const int tid   = threadIdx.x;
    const int lane  = tid & 31;
    const int wave  = tid >> 5;
    const int waveM = wave & 1;
    const int waveN = wave >> 1;
    const int blockM = blockIdx.x * BM;
    const int blockN = blockIdx.y * BN;

    const unsigned ldsB  = (unsigned)(uintptr_t)Bp;
    const unsigned ldsA0 = (unsigned)(uintptr_t)At;
    const unsigned ldsA1 = ldsA0 + (unsigned)(BM * APAD * sizeof(_Float16));

    v8f acc[4][2];
#pragma unroll
    for (int i = 0; i < 4; ++i)
#pragma unroll
        for (int j = 0; j < 2; ++j) acc[i][j] = zero_v8f();

    if (wave == 0) {
        // Whole B panel: 128 weight rows (output cols) x full K=512.
        tdm_load_2d(ldsB, Wt + (size_t)blockN * DHID,
                    /*tile_k=*/DHID, /*tile_rows=*/BN,
                    /*tensor_k=*/DHID, /*tensor_rows=*/DHID,
                    /*row_stride=*/DHID,
                    /*pad_interval=*/7 /*256 dwords*/, /*pad_amount=*/3 /*4 dwords*/);
        // First A tile.
        tdm_load_2d(ldsA0, X + (size_t)blockM * DHID,
                    /*tile_k=*/BK, /*tile_rows=*/BM,
                    /*tensor_k=*/DHID, /*tensor_rows=*/(unsigned)M,
                    /*row_stride=*/DHID,
                    /*pad_interval=*/3 /*16 dwords*/, /*pad_amount=*/3 /*4 dwords*/);
        __builtin_amdgcn_s_wait_tensorcnt(0);
    }
    __syncthreads();

    const int rr = lane & 15;
    const int kb = (lane >> 4) * 8;

    for (int kk = 0; kk < DHID / BK; ++kk) {
        const int k0 = kk * BK;
        // Prefetch next A tile into the other buffer while we compute.
        if (wave == 0 && kk + 1 < DHID / BK) {
            tdm_load_2d(((kk + 1) & 1) ? ldsA1 : ldsA0,
                        X + (size_t)blockM * DHID + (k0 + BK),
                        BK, BM, DHID, (unsigned)M, DHID, 3, 3);
        }
        const _Float16* Ab = At + (size_t)(kk & 1) * (BM * APAD);

        Frag16 fa[4], fb[2];
#pragma unroll
        for (int mt = 0; mt < 4; ++mt) {
            const int r = waveM * 64 + mt * 16 + rr;
            fa[mt].h[0] = *(const v8h*)&Ab[r * APAD + kb];
            fa[mt].h[1] = *(const v8h*)&Ab[r * APAD + kb + 16];
        }
#pragma unroll
        for (int nt = 0; nt < 2; ++nt) {
            const int c = waveN * 32 + nt * 16 + rr;
            fb[nt].h[0] = *(const v8h*)&Bp[c * BPAD + k0 + kb];
            fb[nt].h[1] = *(const v8h*)&Bp[c * BPAD + k0 + kb + 16];
        }
#pragma unroll
        for (int mt = 0; mt < 4; ++mt)
#pragma unroll
            for (int nt = 0; nt < 2; ++nt)
                acc[mt][nt] = __builtin_amdgcn_wmma_f32_16x16x32_f16(
                    false, fa[mt].v, false, fb[nt].v, (short)0, acc[mt][nt], false, false);

        if (wave == 0 && kk + 1 < DHID / BK) __builtin_amdgcn_s_wait_tensorcnt(0);
        __syncthreads();
    }

    // ---- epilogue: bias + activation + store ----
    const int rhi = (lane >> 4) * 8;
#pragma unroll
    for (int mt = 0; mt < 4; ++mt) {
#pragma unroll
        for (int nt = 0; nt < 2; ++nt) {
            const int col = blockN + waveN * 32 + nt * 16 + rr;
            const float bv = bias[col];
#pragma unroll
            for (int v = 0; v < 8; ++v) {
                const int row = blockM + waveM * 64 + mt * 16 + rhi + v;
                if (row < M) {
                    float r = acc[mt][nt][v] + bv;
                    if (ACT == 1) r = fmaxf(r, 0.0f);
                    else if (ACT == 2) r = gelu_exact(r);
                    if (OUT_F16)
                        ((_Float16*)Yout)[(size_t)row * DHID + col] = (_Float16)r;
                    else
                        ((float*)Yout)[(size_t)row * DHID + col] = r;
                }
            }
        }
    }
}

// ---------------------------------------------------------------------------
// z = h  (fp32 copy, grid-stride)
// ---------------------------------------------------------------------------
__global__ __launch_bounds__(256) void copy_f32(const float* __restrict__ src,
                                                float* __restrict__ dst, int n) {
    for (int i = blockIdx.x * 256 + threadIdx.x; i < n; i += gridDim.x * 256)
        dst[i] = src[i];
}

__global__ __launch_bounds__(256) void f32_to_f16(const float* __restrict__ src,
                                                  _Float16* __restrict__ dst, int n) {
    for (int i = blockIdx.x * 256 + threadIdx.x; i < n; i += gridDim.x * 256)
        dst[i] = (_Float16)src[i];
}

// ---------------------------------------------------------------------------
// Edge messages: z[dst] += relu(h[src] + e). One block per edge, 2 elems/thread.
// unsafeAtomicAdd -> native global_atomic_add_f32 (bandwidth-bound phase).
// ---------------------------------------------------------------------------
__global__ __launch_bounds__(256) void edge_msg(const int* __restrict__ ei,
                                                const float* __restrict__ h,
                                                const _Float16* __restrict__ e,
                                                float* __restrict__ z, int E_) {
    const int eid = blockIdx.x;
    const int t = threadIdx.x;
    const int src = ei[eid];
    const int dst = ei[E_ + eid];
    const size_t eb = (size_t)eid * DHID;
    const size_t sb = (size_t)src * DHID;
    const size_t db = (size_t)dst * DHID;
    const float m0 = fmaxf(h[sb + t]       + (float)e[eb + t],       0.0f);
    const float m1 = fmaxf(h[sb + t + 256] + (float)e[eb + t + 256], 0.0f);
    unsafeAtomicAdd(&z[db + t],       m0);
    unsafeAtomicAdd(&z[db + t + 256], m1);
}

// ---------------------------------------------------------------------------
// h = LayerNorm(gelu(u) + h)  (in-place on h). One block per row.
// ---------------------------------------------------------------------------
__global__ __launch_bounds__(256) void post_ln(const float* __restrict__ u,
                                               float* __restrict__ h,
                                               const float* __restrict__ g,
                                               const float* __restrict__ b) {
    const int row = blockIdx.x;
    const int t = threadIdx.x;
    const size_t base = (size_t)row * DHID;
    const float x0 = gelu_exact(u[base + t])       + h[base + t];
    const float x1 = gelu_exact(u[base + t + 256]) + h[base + t + 256];
    __shared__ float red[256];
    red[t] = x0 + x1;
    __syncthreads();
    for (int o = 128; o > 0; o >>= 1) {
        if (t < o) red[t] += red[t + o];
        __syncthreads();
    }
    const float mean = red[0] * (1.0f / 512.0f);
    __syncthreads();
    const float d0 = x0 - mean, d1 = x1 - mean;
    red[t] = d0 * d0 + d1 * d1;
    __syncthreads();
    for (int o = 128; o > 0; o >>= 1) {
        if (t < o) red[t] += red[t + o];
        __syncthreads();
    }
    const float rstd = rsqrtf(red[0] * (1.0f / 512.0f) + 1e-5f);
    h[base + t]       = d0 * rstd * g[t] + b[t];
    h[base + t + 256] = d1 * rstd * g[t + 256] + b[t + 256];
}

// ---------------------------------------------------------------------------
// Host-side orchestration
// ---------------------------------------------------------------------------
extern "C" void kernel_launch(void* const* d_in, const int* in_sizes, int n_in,
                              void* d_out, int out_size, void* d_ws, size_t ws_size,
                              hipStream_t stream) {
    const int Nn = 50000, Ee = 160000;
    const size_t SZ = (size_t)DHID * DHID;

    const int*   x_idx    = (const int*)d_in[0];
    const int*   edg_attr = (const int*)d_in[1];
    const int*   edg_idx  = (const int*)d_in[2];
    const float* atom_emb = (const float*)d_in[3];
    const float* atom_g   = (const float*)d_in[4];
    const float* atom_bb  = (const float*)d_in[5];
    const float* atom_w1  = (const float*)d_in[6];
    const float* atom_b1  = (const float*)d_in[7];
    const float* atom_w2  = (const float*)d_in[8];
    const float* atom_b2  = (const float*)d_in[9];
    const float* bond_emb = (const float*)d_in[10];
    const float* bond_g   = (const float*)d_in[11];
    const float* bond_bb  = (const float*)d_in[12];
    const float* bond_w1  = (const float*)d_in[13];
    const float* bond_b1  = (const float*)d_in[14];
    const float* bond_w2  = (const float*)d_in[15];
    const float* bond_b2  = (const float*)d_in[16];
    const float* conv_w1  = (const float*)d_in[17];
    const float* conv_b1  = (const float*)d_in[18];
    const float* conv_w2  = (const float*)d_in[19];
    const float* conv_b2  = (const float*)d_in[20];
    const float* ln_g     = (const float*)d_in[21];
    const float* ln_b     = (const float*)d_in[22];

    float* h = (float*)d_out;

    char* ws = (char*)d_ws;
    size_t off = 0;
    auto carve = [&](size_t bytes) -> void* {
        void* p = ws + off;
        off += (bytes + 255) & ~(size_t)255;
        return p;
    };
    _Float16* wt   = (_Float16*)carve(12 * SZ * sizeof(_Float16));
    _Float16* e16  = (_Float16*)carve((size_t)Ee * DHID * sizeof(_Float16));
    _Float16* gin  = (_Float16*)carve((size_t)Ee * DHID * sizeof(_Float16));
    _Float16* gmid = (_Float16*)carve((size_t)Ee * DHID * sizeof(_Float16));
    float*    z    = (float*)carve((size_t)Nn * DHID * sizeof(float));
    float*    u    = (float*)carve((size_t)Nn * DHID * sizeof(float));
    (void)ws_size; (void)n_in; (void)in_sizes; (void)out_size;

    _Float16* wtA1 = wt + 0 * SZ;
    _Float16* wtA2 = wt + 1 * SZ;
    _Float16* wtB1 = wt + 2 * SZ;
    _Float16* wtB2 = wt + 3 * SZ;
    _Float16* wtC1 = wt + 4 * SZ;
    _Float16* wtC2 = wt + 8 * SZ;

    transpose_w_f16<<<1024, 256, 0, stream>>>(atom_w1, wtA1);
    transpose_w_f16<<<1024, 256, 0, stream>>>(atom_w2, wtA2);
    transpose_w_f16<<<1024, 256, 0, stream>>>(bond_w1, wtB1);
    transpose_w_f16<<<1024, 256, 0, stream>>>(bond_w2, wtB2);
    for (int l = 0; l < 4; ++l) {
        transpose_w_f16<<<1024, 256, 0, stream>>>(conv_w1 + (size_t)l * SZ, wtC1 + (size_t)l * SZ);
        transpose_w_f16<<<1024, 256, 0, stream>>>(conv_w2 + (size_t)l * SZ, wtC2 + (size_t)l * SZ);
    }

    const dim3 gN((Nn + BM - 1) / BM, DHID / BN);
    const dim3 gE((Ee + BM - 1) / BM, DHID / BN);
    const int nTot = Nn * DHID;
    const size_t GEMM_LDS = ((size_t)BN * BPAD + 2 * (size_t)BM * APAD) * sizeof(_Float16); // 150 KB

    // ---- atom encoder ----
    embed_ln_f16<<<Nn, 256, 0, stream>>>(x_idx, atom_emb, atom_g, atom_bb, gin, 9, 128);
    gemm_wmma<2, true ><<<gN, 256, GEMM_LDS, stream>>>(gin,  wtA1, atom_b1, gmid, Nn);
    gemm_wmma<0, false><<<gN, 256, GEMM_LDS, stream>>>(gmid, wtA2, atom_b2, h,    Nn);

    // ---- bond encoder ----
    embed_ln_f16<<<Ee, 256, 0, stream>>>(edg_attr, bond_emb, bond_g, bond_bb, gin, 3, 8);
    gemm_wmma<2, true><<<gE, 256, GEMM_LDS, stream>>>(gin,  wtB1, bond_b1, gmid, Ee);
    gemm_wmma<0, true><<<gE, 256, GEMM_LDS, stream>>>(gmid, wtB2, bond_b2, e16,  Ee);

    // ---- GINE layers ----
    for (int l = 0; l < 4; ++l) {
        copy_f32<<<4096, 256, 0, stream>>>(h, z, nTot);
        edge_msg<<<Ee, 256, 0, stream>>>(edg_idx, h, e16, z, Ee);
        f32_to_f16<<<4096, 256, 0, stream>>>(z, gin, nTot);
        gemm_wmma<1, true ><<<gN, 256, GEMM_LDS, stream>>>(gin,  wtC1 + (size_t)l * SZ,
                                                           conv_b1 + (size_t)l * DHID, gmid, Nn);
        gemm_wmma<0, false><<<gN, 256, GEMM_LDS, stream>>>(gmid, wtC2 + (size_t)l * SZ,
                                                           conv_b2 + (size_t)l * DHID, u,    Nn);
        post_ln<<<Nn, 256, 0, stream>>>(u, h, ln_g + (size_t)l * DHID, ln_b + (size_t)l * DHID);
    }
}
